// MegaLLM_70128226009717
// MI455X (gfx1250) — matrix operand order repaired and checked by reading, not compiled, not run on hardware
//
#include <hip/hip_runtime.h>
#include <math.h>

// ---------------------------------------------------------------------------
// MegaLLM forward for gfx1250 (MI455X).
//  - All GEMMs: v_wmma_f32_16x16x32_bf16, fp32 accumulate, LDS-tiled,
//    register double-buffered global loads, global_prefetch_b8 hints.
//  - Attention: flash-style bf16 WMMA (QK^T and P@V), online softmax with
//    cross-lane reductions in the WMMA C-fragment layout.
// ---------------------------------------------------------------------------

typedef __attribute__((ext_vector_type(16))) __bf16 v16bf;
typedef __attribute__((ext_vector_type(8)))  float  v8f;

union Frag16 { v16bf v; unsigned int u[8]; };

#define BM 128
#define BN 128
#define BK 32
#define ASTR 34   // padded LDS row strides (halves) -- even for 32-bit reads
#define BSTR 34

// ---------------------------------------------------------------------------
// Embedding gather: x[row,:] = embed[ids[row],:]
// ---------------------------------------------------------------------------
__global__ void embed_kernel(const int* __restrict__ ids,
                             const float* __restrict__ embed,
                             float* __restrict__ x, int D) {
    int row = blockIdx.x;
    int id = ids[row];
    const float4* src = (const float4*)(embed + (size_t)id * D);
    float4* dst = (float4*)(x + (size_t)row * D);
    for (int i = threadIdx.x; i < D / 4; i += blockDim.x) dst[i] = src[i];
}

// ---------------------------------------------------------------------------
// RMSNorm: out_bf16[row,:] = x * rsqrt(mean(x^2)+eps) * g
// ---------------------------------------------------------------------------
__global__ void rmsnorm_kernel(const float* __restrict__ x,
                               const float* __restrict__ g,
                               __bf16* __restrict__ out, int D) {
    __shared__ float red[256];
    int row = blockIdx.x;
    int tid = threadIdx.x;
    const float* xr = x + (size_t)row * D;
    float s = 0.0f;
    for (int i = tid; i < D; i += 256) { float v = xr[i]; s += v * v; }
    red[tid] = s;
    __syncthreads();
    for (int st = 128; st > 0; st >>= 1) {
        if (tid < st) red[tid] += red[tid + st];
        __syncthreads();
    }
    float rstd = rsqrtf(red[0] / (float)D + 1e-6f);
    __bf16* orow = out + (size_t)row * D;
    for (int i = tid; i < D; i += 256)
        orow[i] = (__bf16)(xr[i] * rstd * g[i]);
}

// ---------------------------------------------------------------------------
// Tiled WMMA GEMM: C[M,N] = A_bf16[M,K] @ W_f32[K,N] + bias
//  MODE 0: C fp32 = acc + bias
//  MODE 1: C fp32 = resid + acc + bias
//  MODE 2: C bf16 = silu(acc + bias)
//  MODE 3: C bf16 = acc + bias
// 256 threads = 8 waves (4 M x 2 N); wave tile 32x64 = 2x4 16x16 fragments.
// ---------------------------------------------------------------------------
template <int MODE>
__global__ __launch_bounds__(256) void gemm_bf16_kernel(
    const __bf16* __restrict__ A, const float* __restrict__ W,
    const float* __restrict__ bias, void* __restrict__ Cv,
    const float* __restrict__ resid, int M, int N, int K) {
    __shared__ __align__(16) __bf16 Asm[BM * ASTR];
    __shared__ __align__(16) __bf16 Bsm[BN * BSTR];   // transposed: [n][k]

    int tid  = threadIdx.x;
    int lane = tid & 31;
    int wid  = tid >> 5;
    int wm   = wid >> 1;          // 0..3
    int wn   = wid & 1;           // 0..1
    int bm   = blockIdx.y * BM;
    int bn   = blockIdx.x * BN;
    int lg   = lane >> 4;
    int lm   = lane & 15;

    // global-load assignments
    int r0 = tid >> 2;            // 0..63 (A rows r0, r0+64)
    int kh = (tid & 3) * 8;       // A halves offset
    int kk = tid >> 3;            // 0..31 (W k row)
    int n0 = (tid & 7) * 16;      // W n offset (16 floats)

    const __bf16* Abase = A + (size_t)bm * K;

    v8f acc[2][4];
    #pragma unroll
    for (int i = 0; i < 2; i++)
        #pragma unroll
        for (int j = 0; j < 4; j++)
            #pragma unroll
            for (int r = 0; r < 8; r++) acc[i][j][r] = 0.0f;

    // prologue: stage tile kt=0 into registers
    uint4 aR0 = *(const uint4*)(Abase + (size_t)r0 * K + kh);
    uint4 aR1 = *(const uint4*)(Abase + (size_t)(r0 + 64) * K + kh);
    float4 wR[4];
    {
        const float* wp = W + (size_t)kk * N + bn + n0;
        #pragma unroll
        for (int j = 0; j < 4; j++) wR[j] = ((const float4*)wp)[j];
    }

    for (int kt = 0; kt < K; kt += BK) {
        // ---- registers -> LDS ----
        {
            unsigned int* d0 = (unsigned int*)&Asm[r0 * ASTR + kh];
            d0[0] = aR0.x; d0[1] = aR0.y; d0[2] = aR0.z; d0[3] = aR0.w;
            unsigned int* d1 = (unsigned int*)&Asm[(r0 + 64) * ASTR + kh];
            d1[0] = aR1.x; d1[1] = aR1.y; d1[2] = aR1.z; d1[3] = aR1.w;
            #pragma unroll
            for (int j = 0; j < 4; j++) {
                float4 f = wR[j];
                int nb = n0 + j * 4;
                Bsm[(nb + 0) * BSTR + kk] = (__bf16)f.x;
                Bsm[(nb + 1) * BSTR + kk] = (__bf16)f.y;
                Bsm[(nb + 2) * BSTR + kk] = (__bf16)f.z;
                Bsm[(nb + 3) * BSTR + kk] = (__bf16)f.w;
            }
        }
        __syncthreads();

        // ---- issue next tile's global loads (overlap with WMMA below) ----
        if (kt + BK < K) {
            int kn = kt + BK;
            aR0 = *(const uint4*)(Abase + (size_t)r0 * K + kn + kh);
            aR1 = *(const uint4*)(Abase + (size_t)(r0 + 64) * K + kn + kh);
            const float* wp = W + (size_t)(kn + kk) * N + bn + n0;
            #pragma unroll
            for (int j = 0; j < 4; j++) wR[j] = ((const float4*)wp)[j];
            if (kt + 2 * BK < K) {   // gfx1250 global_prefetch_b8 hints
                __builtin_prefetch(W + (size_t)(kt + 2 * BK + kk) * N + bn + n0, 0, 1);
                __builtin_prefetch(Abase + (size_t)r0 * K + kt + 2 * BK + kh, 0, 1);
            }
        }

        // ---- fragments from LDS ----
        Frag16 a[2], b[4];
        #pragma unroll
        for (int fm = 0; fm < 2; ++fm) {
            const unsigned int* p =
                (const unsigned int*)&Asm[(wm * 32 + fm * 16 + lm) * ASTR];
            int kb2 = lg * 4;
            #pragma unroll
            for (int i = 0; i < 4; i++) {
                a[fm].u[i]     = p[kb2 + i];
                a[fm].u[4 + i] = p[8 + kb2 + i];
            }
        }
        #pragma unroll
        for (int fn = 0; fn < 4; ++fn) {
            const unsigned int* p =
                (const unsigned int*)&Bsm[(wn * 64 + fn * 16 + lm) * BSTR];
            #pragma unroll
            for (int i = 0; i < 8; i++) b[fn].u[i] = p[(i >> 2) * 8 + lg * 4 + (i & 3)];
        }

        #pragma unroll
        for (int fm = 0; fm < 2; ++fm)
            #pragma unroll
            for (int fn = 0; fn < 4; ++fn)
                acc[fm][fn] = __builtin_amdgcn_wmma_f32_16x16x32_bf16(
                    false, a[fm].v, false, b[fn].v,
                    (short)0, acc[fm][fn], false, false);
        __syncthreads();
    }

    // ---- epilogue ----
    #pragma unroll
    for (int fm = 0; fm < 2; ++fm) {
        int mbase = bm + wm * 32 + fm * 16 + lg * 8;
        #pragma unroll
        for (int fn = 0; fn < 4; ++fn) {
            int n = bn + wn * 64 + fn * 16 + lm;
            float bv = bias[n];
            #pragma unroll
            for (int r = 0; r < 8; r++) {
                size_t idx = (size_t)(mbase + r) * N + n;
                float v = acc[fm][fn][r] + bv;
                if (MODE == 0) {
                    ((float*)Cv)[idx] = v;
                } else if (MODE == 1) {
                    ((float*)Cv)[idx] = resid[idx] + v;
                } else if (MODE == 2) {
                    ((__bf16*)Cv)[idx] = (__bf16)(v / (1.0f + __expf(-v)));
                } else {
                    ((__bf16*)Cv)[idx] = (__bf16)v;
                }
            }
        }
    }
}

// ---------------------------------------------------------------------------
// Flash attention, bf16 WMMA. Block = 256 threads = 8 waves; each wave owns
// 16 queries; block covers 128 queries of one (batch, head). Key tiles of 32
// staged in LDS (K as [key][hd], V transposed as [hd][key]).
// C-fragment layout: lane l holds column n=l&15, rows m=r+8*(l>>4) -> per-row
// softmax stats reduce across the 16 lanes sharing a row via __shfl_xor.
// ---------------------------------------------------------------------------
#define KSTR 66
#define VSTR 34
#define PSTR 34

__global__ __launch_bounds__(256) void flash_attn_kernel(
    const __bf16* __restrict__ Qg, const __bf16* __restrict__ Kg,
    const __bf16* __restrict__ Vg, __bf16* __restrict__ Og, int S) {
    __shared__ __align__(16) __bf16 Ksm[32 * KSTR];
    __shared__ __align__(16) __bf16 Vsm[64 * VSTR];
    __shared__ __align__(16) __bf16 Psm[8 * 16 * PSTR];

    int tid = threadIdx.x, lane = tid & 31, wid = tid >> 5;
    int lg = lane >> 4, lm = lane & 15;
    int h = blockIdx.y, b = blockIdx.z;
    int QB = blockIdx.x * 128;
    int qw = QB + wid * 16;                    // wave's first query row

    // ---- Q fragments (16 queries x 64 hd = 2 A-frags) ----
    Frag16 qf[2];
    {
        const __bf16* qrow = Qg + ((size_t)(b * S + qw + lm)) * 1024 + h * 64;
        #pragma unroll
        for (int c = 0; c < 2; ++c) {
            int base = c * 32 + lg * 8;
            #pragma unroll
            for (int i = 0; i < 4; i++) {
                qf[c].u[i]     = *(const unsigned int*)(qrow + base + 2 * i);
                qf[c].u[4 + i] = *(const unsigned int*)(qrow + base + 16 + 2 * i);
            }
        }
    }

    float rmax[8], rsum[8];
    v8f oacc[4];
    #pragma unroll
    for (int r = 0; r < 8; r++) { rmax[r] = -1e30f; rsum[r] = 0.0f; }
    #pragma unroll
    for (int fn = 0; fn < 4; fn++)
        #pragma unroll
        for (int r = 0; r < 8; r++) oacc[fn][r] = 0.0f;

    int ntiles = (QB + 128) / 32;
    for (int t = 0; t < ntiles; ++t) {
        int kb = t * 32;
        __syncthreads();
        // ---- stage K tile [32][64] and V^T tile [64][32] ----
        {
            int key = tid >> 3, hs = (tid & 7) * 8;
            size_t g = ((size_t)(b * S + kb + key)) * 1024 + h * 64 + hs;
            uint4 kd = *(const uint4*)(Kg + g);
            unsigned int* kdst = (unsigned int*)&Ksm[key * KSTR + hs];
            kdst[0] = kd.x; kdst[1] = kd.y; kdst[2] = kd.z; kdst[3] = kd.w;
            uint4 vd = *(const uint4*)(Vg + g);
            const __bf16* vh = (const __bf16*)&vd;
            #pragma unroll
            for (int j = 0; j < 8; j++) Vsm[(hs + j) * VSTR + key] = vh[j];
        }
        __syncthreads();

        if (kb <= qw + 15) {                   // wave-uniform: EXEC stays full
            // ---- scores: S = Q @ K^T (2 key groups x 2 hd chunks) ----
            v8f s[2];
            #pragma unroll
            for (int f = 0; f < 2; f++)
                #pragma unroll
                for (int r = 0; r < 8; r++) s[f][r] = 0.0f;
            #pragma unroll
            for (int f = 0; f < 2; ++f) {
                #pragma unroll
                for (int c = 0; c < 2; ++c) {
                    Frag16 kf;
                    const unsigned int* p =
                        (const unsigned int*)&Ksm[(f * 16 + lm) * KSTR];
                    #pragma unroll
                    for (int i = 0; i < 8; i++)
                        kf.u[i] = p[c * 16 + (i >> 2) * 8 + lg * 4 + (i & 3)];
                    s[f] = __builtin_amdgcn_wmma_f32_16x16x32_bf16(
                        false, qf[c].v, false, kf.v, (short)0, s[f],
                        false, false);
                }
            }
            // ---- scale + causal mask ----
            #pragma unroll
            for (int f = 0; f < 2; ++f)
                #pragma unroll
                for (int r = 0; r < 8; ++r) {
                    float v = s[f][r] * 0.125f;
                    int qy = qw + r + 8 * lg;
                    if (kb + f * 16 + lm > qy) v = -1e30f;
                    s[f][r] = v;
                }
            // ---- online softmax update ----
            #pragma unroll
            for (int r = 0; r < 8; ++r) {
                float v = fmaxf(s[0][r], s[1][r]);
                v = fmaxf(v, __shfl_xor(v, 1, 32));
                v = fmaxf(v, __shfl_xor(v, 2, 32));
                v = fmaxf(v, __shfl_xor(v, 4, 32));
                v = fmaxf(v, __shfl_xor(v, 8, 32));
                float nmax = fmaxf(rmax[r], v);
                float p0 = __expf(s[0][r] - nmax);
                float p1 = __expf(s[1][r] - nmax);
                s[0][r] = p0; s[1][r] = p1;
                float ps = p0 + p1;
                ps += __shfl_xor(ps, 1, 32);
                ps += __shfl_xor(ps, 2, 32);
                ps += __shfl_xor(ps, 4, 32);
                ps += __shfl_xor(ps, 8, 32);
                float fct = __expf(rmax[r] - nmax);
                rsum[r] = rsum[r] * fct + ps;
                rmax[r] = nmax;
                #pragma unroll
                for (int fn = 0; fn < 4; ++fn) oacc[fn][r] *= fct;
            }
            // ---- restage P (C-layout -> LDS row-major) ----
            __bf16* pb = &Psm[wid * 16 * PSTR];
            #pragma unroll
            for (int f = 0; f < 2; ++f)
                #pragma unroll
                for (int r = 0; r < 8; ++r)
                    pb[(r + 8 * lg) * PSTR + f * 16 + lm] = (__bf16)s[f][r];
            asm volatile("" ::: "memory");     // keep LDS write->read ordered
            // ---- P as A-fragment ----
            Frag16 pf;
            {
                const unsigned int* pp =
                    (const unsigned int*)&Psm[(wid * 16 + lm) * PSTR];
                #pragma unroll
                for (int i = 0; i < 4; i++) {
                    pf.u[i]     = pp[lg * 4 + i];
                    pf.u[4 + i] = pp[8 + lg * 4 + i];
                }
            }
            // ---- O += P @ V ----
            #pragma unroll
            for (int fn = 0; fn < 4; ++fn) {
                Frag16 vf;
                const unsigned int* vp =
                    (const unsigned int*)&Vsm[(fn * 16 + lm) * VSTR];
                #pragma unroll
                for (int i = 0; i < 8; i++) vf.u[i] = vp[(i >> 2) * 8 + lg * 4 + (i & 3)];
                oacc[fn] = __builtin_amdgcn_wmma_f32_16x16x32_bf16(
                    false, pf.v, false, vf.v, (short)0, oacc[fn],
                    false, false);
            }
        }
    }

    // ---- normalize + write O (bf16) ----
    #pragma unroll
    for (int r = 0; r < 8; ++r) {
        float inv = 1.0f / rsum[r];
        int qy = qw + r + 8 * lg;
        size_t base = ((size_t)(b * S + qy)) * 1024 + h * 64;
        #pragma unroll
        for (int fn = 0; fn < 4; ++fn)
            Og[base + fn * 16 + lm] = (__bf16)(oacc[fn][r] * inv);
    }
}

// ---------------------------------------------------------------------------
// Orchestration
// ---------------------------------------------------------------------------
extern "C" void kernel_launch(void* const* d_in, const int* in_sizes, int n_in,
                              void* d_out, int out_size, void* d_ws, size_t ws_size,
                              hipStream_t stream) {
    const int*   ids    = (const int*)  d_in[0];
    const float* embed  = (const float*)d_in[1];
    const float* g1     = (const float*)d_in[2];
    const float* wq     = (const float*)d_in[3];
    const float* bq     = (const float*)d_in[4];
    const float* wk     = (const float*)d_in[5];
    const float* bk     = (const float*)d_in[6];
    const float* wv     = (const float*)d_in[7];
    const float* bv     = (const float*)d_in[8];
    const float* wo     = (const float*)d_in[9];
    const float* bo     = (const float*)d_in[10];
    const float* g2     = (const float*)d_in[11];
    const float* w1     = (const float*)d_in[12];
    const float* b1     = (const float*)d_in[13];
    const float* w2     = (const float*)d_in[14];
    const float* b2     = (const float*)d_in[15];
    const float* gf     = (const float*)d_in[16];
    const float* head_w = (const float*)d_in[17];
    const float* head_b = (const float*)d_in[18];

    const int Bsz = 2, S = 1024, D = 1024, F = 4096, V = 32000;
    const int Mr = Bsz * S;                    // 2048

    // workspace layout (bytes); total 44 MB
    char* ws = (char*)d_ws;
    float*  x  = (float*) (ws + 0);                           //  8 MB fp32
    __bf16* h  = (__bf16*)(ws + 8u  * 1024 * 1024);           //  4 MB bf16
    __bf16* qb = (__bf16*)(ws + 12u * 1024 * 1024);           //  4 MB bf16
    __bf16* kb = (__bf16*)(ws + 16u * 1024 * 1024);           //  4 MB bf16
    __bf16* vb = (__bf16*)(ws + 20u * 1024 * 1024);           //  4 MB bf16
    __bf16* ob = (__bf16*)(ws + 24u * 1024 * 1024);           //  4 MB bf16
    __bf16* h1 = (__bf16*)(ws + 28u * 1024 * 1024);           // 16 MB bf16
    (void)ws_size; (void)in_sizes; (void)n_in; (void)out_size;

    dim3 blk(256);
    dim3 gD(D / 128, Mr / 128);                // (8, 16)
    dim3 gF(F / 128, Mr / 128);                // (32, 16)
    dim3 gV(V / 128, Mr / 128);                // (250, 16)
    dim3 gAttn(S / 128, 16, Bsz);              // (8, 16, 2)

    embed_kernel<<<Mr, blk, 0, stream>>>(ids, embed, x, D);

    for (int l = 0; l < 2; ++l) {
        const size_t wD = (size_t)l * D * D;
        // ---- attention sublayer ----
        rmsnorm_kernel<<<Mr, blk, 0, stream>>>(x, g1 + l * D, h, D);
        gemm_bf16_kernel<3><<<gD, blk, 0, stream>>>(h, wq + wD, bq + l * D, qb, nullptr, Mr, D, D);
        gemm_bf16_kernel<3><<<gD, blk, 0, stream>>>(h, wk + wD, bk + l * D, kb, nullptr, Mr, D, D);
        gemm_bf16_kernel<3><<<gD, blk, 0, stream>>>(h, wv + wD, bv + l * D, vb, nullptr, Mr, D, D);
        flash_attn_kernel<<<gAttn, blk, 0, stream>>>(qb, kb, vb, ob, S);
        gemm_bf16_kernel<1><<<gD, blk, 0, stream>>>(ob, wo + wD, bo + l * D, x, x, Mr, D, D);
        // ---- FFN sublayer ----
        rmsnorm_kernel<<<Mr, blk, 0, stream>>>(x, g2 + l * D, h, D);
        gemm_bf16_kernel<2><<<gF, blk, 0, stream>>>(h,  w1 + (size_t)l * D * F, b1 + l * F, h1, nullptr, Mr, F, D);
        gemm_bf16_kernel<1><<<gD, blk, 0, stream>>>(h1, w2 + (size_t)l * F * D, b2 + l * D, x,  x,      Mr, D, F);
    }

    // ---- final norm + vocab head ----
    rmsnorm_kernel<<<Mr, blk, 0, stream>>>(x, gf, h, D);
    gemm_bf16_kernel<0><<<gV, blk, 0, stream>>>(h, head_w, head_b, (float*)d_out,
                                                nullptr, Mr, V, D);
}
